// NoParamLayer_10136122819075
// MI455X (gfx1250) — matrix operand order repaired and checked by reading, not compiled, not run on hardware
//
#include <hip/hip_runtime.h>

typedef __attribute__((ext_vector_type(16))) _Float16 v16h;
typedef __attribute__((ext_vector_type(8)))  float    v8f;

#define B_   32
#define LQ_  1024
#define LKV_ 1024
#define D_   512

union V16U { v16h h; uint4 u[2]; _Float16 e[16]; };

#define WMMA_F16(A, Bm, C) \
  __builtin_amdgcn_wmma_f32_16x16x32_f16(false, (A), false, (Bm), (short)0, (C), false, false)

// ---------------------------------------------------------------------------
// KV f32 -> f16 pre-convert (8 elements / thread, b128 in / b128 out)
// ---------------------------------------------------------------------------
__global__ __launch_bounds__(256)
void cvt_f32_to_f16(const float* __restrict__ src, _Float16* __restrict__ dst) {
  const size_t i = ((size_t)blockIdx.x * 256 + threadIdx.x) * 8;
  const float4 f0 = *(const float4*)(src + i);
  const float4 f1 = *(const float4*)(src + i + 4);
  union { _Float16 e[8]; uint4 u; } o;
  o.e[0] = (_Float16)f0.x; o.e[1] = (_Float16)f0.y;
  o.e[2] = (_Float16)f0.z; o.e[3] = (_Float16)f0.w;
  o.e[4] = (_Float16)f1.x; o.e[5] = (_Float16)f1.y;
  o.e[6] = (_Float16)f1.z; o.e[7] = (_Float16)f1.w;
  *(uint4*)(dst + i) = o.u;
}

// ---------------------------------------------------------------------------
// Main fused kernel: one block = batch b, 32 query rows, 8 waves.
// Each wave: phase A -> 2 q-tiles x 8 kv-tiles (B operand shared),
//            phase B -> 2 q-tiles x 4 d-tiles  (B operand shared).
// ---------------------------------------------------------------------------
__global__ __launch_bounds__(256)
void attn_fused32(const float* __restrict__ Q, const _Float16* __restrict__ KVh,
                  float* __restrict__ O, float* __restrict__ ATT) {
  // LDS map (98.25KB):
  //  [0,32K)       phase A: Qh f16 [32][512]  / phase B: KVT f16 [512][32] swizzled
  //  [32K,96K)     Ph f16 [32][1024]
  //  [96K,98.25K)  softmax reduce scratch (phase A only)
  __shared__ __align__(16) unsigned char smem[100608];
  _Float16* Qh  = (_Float16*)smem;
  _Float16* KVT = (_Float16*)smem;
  _Float16* Ph  = (_Float16*)(smem + 32768);
  float*    red = (float*)(smem + 98304);
  float* redmax = red;            // [8][32]
  float* redsum = red + 256;      // [8][32]
  float* gmax   = red + 512;      // [32]
  float* gsum   = red + 544;      // [32]

  const int tid  = threadIdx.x;
  const int w    = tid >> 5;
  const int lane = tid & 31;
  const int hs   = lane >> 4;
  const int l15  = lane & 15;

  const int b  = blockIdx.y;
  const int q0 = blockIdx.x << 5;           // 32 q rows / block

  const float*    Qb   = Q   + ((size_t)b * LQ_ + q0) * D_;
  const _Float16* KVhb = KVh + (size_t)b * LKV_ * D_;

  // ---- Stage Q strip (32x512) as f16 ----
  for (int e = tid; e < 32 * D_; e += 256) Qh[e] = (_Float16)Qb[e];
  __syncthreads();

  // ---- Phase A: S = Q * KV^T  (32 x 1024 strip in registers) ----
  v8f acc[16] = {};                          // [qt*8 + t]
  for (int dk = 0; dk < D_ / 32; ++dk) {
    V16U a0, a1;
    const _Float16* ap0 = Qh + l15 * D_ + dk * 32 + hs * 8;
    const _Float16* ap1 = ap0 + 16 * D_;
    a0.u[0] = *(const uint4*)(ap0);  a0.u[1] = *(const uint4*)(ap0 + 16);
    a1.u[0] = *(const uint4*)(ap1);  a1.u[1] = *(const uint4*)(ap1 + 16);
#pragma unroll
    for (int t = 0; t < 8; ++t) {
      const int kvrow = (w * 8 + t) * 16 + l15;
      const _Float16* bp = KVhb + (size_t)kvrow * D_ + dk * 32 + hs * 8;
      V16U bm;
      bm.u[0] = *(const uint4*)(bp);
      bm.u[1] = *(const uint4*)(bp + 16);
      acc[t]     = WMMA_F16(a0.h, bm.h, acc[t]);
      acc[8 + t] = WMMA_F16(a1.h, bm.h, acc[8 + t]);
    }
  }

  // ---- Softmax over 32 rows (row = qt*16 + hs*8 + r, col striped per wave) --
  float pmax[16];
#pragma unroll
  for (int qt = 0; qt < 2; ++qt)
#pragma unroll
    for (int r = 0; r < 8; ++r) {
      float m = acc[qt * 8][r];
#pragma unroll
      for (int t = 1; t < 8; ++t) m = fmaxf(m, acc[qt * 8 + t][r]);
      pmax[qt * 8 + r] = m;
    }
#pragma unroll
  for (int s = 1; s < 16; s <<= 1)
#pragma unroll
    for (int i = 0; i < 16; ++i)
      pmax[i] = fmaxf(pmax[i], __shfl_xor(pmax[i], s, 32));
  if (l15 == 0) {
#pragma unroll
    for (int qt = 0; qt < 2; ++qt)
#pragma unroll
      for (int r = 0; r < 8; ++r)
        redmax[w * 32 + qt * 16 + hs * 8 + r] = pmax[qt * 8 + r];
  }
  __syncthreads();
  if (tid < 32) {
    float m = redmax[tid];
#pragma unroll
    for (int w2 = 1; w2 < 8; ++w2) m = fmaxf(m, redmax[w2 * 32 + tid]);
    gmax[tid] = m;
  }
  __syncthreads();

  float psum[16];
#pragma unroll
  for (int qt = 0; qt < 2; ++qt)
#pragma unroll
    for (int r = 0; r < 8; ++r) {
      const float gm = gmax[qt * 16 + hs * 8 + r];
      float s = 0.f;
#pragma unroll
      for (int t = 0; t < 8; ++t) {
        const float ev = __expf(acc[qt * 8 + t][r] - gm);
        acc[qt * 8 + t][r] = ev;
        s += ev;
      }
      psum[qt * 8 + r] = s;
    }
#pragma unroll
  for (int s = 1; s < 16; s <<= 1)
#pragma unroll
    for (int i = 0; i < 16; ++i) psum[i] += __shfl_xor(psum[i], s, 32);
  if (l15 == 0) {
#pragma unroll
    for (int qt = 0; qt < 2; ++qt)
#pragma unroll
      for (int r = 0; r < 8; ++r)
        redsum[w * 32 + qt * 16 + hs * 8 + r] = psum[qt * 8 + r];
  }
  __syncthreads();
  if (tid < 32) {
    float s = redsum[tid];
#pragma unroll
    for (int w2 = 1; w2 < 8; ++w2) s += redsum[w2 * 32 + tid];
    gsum[tid] = s;
  }
  __syncthreads();

  // ---- Normalize; write attn (f32 output) + Ph (f16 for phase B) ----
  float* ATTb = ATT + ((size_t)b * LQ_ + q0) * LKV_;
#pragma unroll
  for (int qt = 0; qt < 2; ++qt)
#pragma unroll
    for (int r = 0; r < 8; ++r) {
      const int grow = qt * 16 + hs * 8 + r;
      const float inv = 1.0f / gsum[grow];
#pragma unroll
      for (int t = 0; t < 8; ++t) {
        const int col = (w * 8 + t) * 16 + l15;
        const float p = acc[qt * 8 + t][r] * inv;
        ATTb[(size_t)grow * LKV_ + col] = p;
        Ph[grow * LKV_ + col] = (_Float16)p;
      }
    }
  __syncthreads();   // Ph complete; Qh/red dead -> KVT reuses region

  // ---- Phase B: O = P * KV ----
  v8f accO[8] = {};                          // [qt*4 + dt]
  for (int kc = 0; kc < LKV_ / 32; ++kc) {
    // Stage 32(kv) x 512(d) f16 chunk transposed + XOR-swizzled: KVT[d][kv']
    for (int e = tid; e < 32 * (D_ / 2); e += 256) {
      const int kv = e >> 8;
      const int d0 = (e & 255) << 1;
      union { unsigned int u; _Float16 h[2]; } v;
      v.u = *(const unsigned int*)(KVhb + (size_t)(kc * 32 + kv) * D_ + d0);
      const int q8 = ((kv >> 3) ^ ((d0 >> 2) & 3)) << 3;
      KVT[d0 * 32 + q8 + (kv & 7)]        = v.h[0];
      KVT[(d0 + 1) * 32 + q8 + (kv & 7)]  = v.h[1];
    }
    __syncthreads();

    V16U a0, a1;
    const _Float16* ap0 = Ph + l15 * LKV_ + kc * 32 + hs * 8;
    const _Float16* ap1 = ap0 + 16 * LKV_;
    a0.u[0] = *(const uint4*)(ap0);  a0.u[1] = *(const uint4*)(ap0 + 16);
    a1.u[0] = *(const uint4*)(ap1);  a1.u[1] = *(const uint4*)(ap1 + 16);
#pragma unroll
    for (int dt = 0; dt < 4; ++dt) {
      const int dcol = (w * 4 + dt) * 16 + l15;
      const int sw = (dcol >> 2) & 3;
      const uint4* rp = (const uint4*)(KVT + dcol * 32);
      V16U bm;
      bm.u[0] = rp[hs ^ sw];
      bm.u[1] = rp[(hs + 2) ^ sw];
      accO[dt]     = WMMA_F16(a0.h, bm.h, accO[dt]);
      accO[4 + dt] = WMMA_F16(a1.h, bm.h, accO[4 + dt]);
    }
    __syncthreads();
  }

  float* Ob = O + ((size_t)b * LQ_ + q0) * D_;
#pragma unroll
  for (int qt = 0; qt < 2; ++qt)
#pragma unroll
    for (int dt = 0; dt < 4; ++dt)
#pragma unroll
      for (int r = 0; r < 8; ++r)
        Ob[(size_t)(qt * 16 + hs * 8 + r) * D_ + (w * 4 + dt) * 16 + l15] =
            accO[qt * 4 + dt][r];
}

// ---------------------------------------------------------------------------
// Fallback (no workspace): previous fused kernel, f32 KV converted inline.
// ---------------------------------------------------------------------------
__global__ __launch_bounds__(256)
void attn_fused(const float* __restrict__ Q, const float* __restrict__ KV,
                float* __restrict__ O, float* __restrict__ ATT) {
  __shared__ __align__(16) unsigned char smem[65536];
  _Float16* Qh  = (_Float16*)smem;
  float*    red = (float*)(smem + 16384);
  _Float16* KVT = (_Float16*)smem;
  _Float16* Ph  = (_Float16*)(smem + 32768);
  float* redmax = red;
  float* redsum = red + 128;
  float* gmax   = red + 256;
  float* gsum   = red + 272;

  const int tid  = threadIdx.x;
  const int w    = tid >> 5;
  const int lane = tid & 31;
  const int hs   = lane >> 4;
  const int l15  = lane & 15;
  const int b  = blockIdx.y;
  const int q0 = blockIdx.x << 4;

  const float* Qb  = Q  + ((size_t)b * LQ_ + q0) * D_;
  const float* KVb = KV + (size_t)b * LKV_ * D_;

  for (int e = tid; e < 16 * D_; e += 256) Qh[e] = (_Float16)Qb[e];
  __syncthreads();

  v8f acc[8] = {};
  for (int dk = 0; dk < D_ / 32; ++dk) {
    V16U a;
    const _Float16* ap = Qh + l15 * D_ + dk * 32 + hs * 8;
    a.u[0] = *(const uint4*)(ap);
    a.u[1] = *(const uint4*)(ap + 16);
#pragma unroll
    for (int t = 0; t < 8; ++t) {
      const int kvrow = (w * 8 + t) * 16 + l15;
      const float* sf = KVb + (size_t)kvrow * D_ + dk * 32 + hs * 8;
      V16U bm;
#pragma unroll
      for (int j = 0; j < 16; ++j) bm.e[j] = (_Float16)sf[j + (j & 8)];
      acc[t] = WMMA_F16(a.h, bm.h, acc[t]);
    }
  }

  float pmax[8];
#pragma unroll
  for (int r = 0; r < 8; ++r) {
    float m = acc[0][r];
#pragma unroll
    for (int t = 1; t < 8; ++t) m = fmaxf(m, acc[t][r]);
    pmax[r] = m;
  }
#pragma unroll
  for (int s = 1; s < 16; s <<= 1)
#pragma unroll
    for (int r = 0; r < 8; ++r)
      pmax[r] = fmaxf(pmax[r], __shfl_xor(pmax[r], s, 32));
  if (l15 == 0) {
#pragma unroll
    for (int r = 0; r < 8; ++r) redmax[w * 16 + hs * 8 + r] = pmax[r];
  }
  __syncthreads();
  if (tid < 16) {
    float m = redmax[tid];
#pragma unroll
    for (int w2 = 1; w2 < 8; ++w2) m = fmaxf(m, redmax[w2 * 16 + tid]);
    gmax[tid] = m;
  }
  __syncthreads();

  float psum[8];
#pragma unroll
  for (int r = 0; r < 8; ++r) {
    const float gm = gmax[hs * 8 + r];
    float s = 0.f;
#pragma unroll
    for (int t = 0; t < 8; ++t) {
      const float ev = __expf(acc[t][r] - gm);
      acc[t][r] = ev;
      s += ev;
    }
    psum[r] = s;
  }
#pragma unroll
  for (int s = 1; s < 16; s <<= 1)
#pragma unroll
    for (int r = 0; r < 8; ++r) psum[r] += __shfl_xor(psum[r], s, 32);
  if (l15 == 0) {
#pragma unroll
    for (int r = 0; r < 8; ++r) redsum[w * 16 + hs * 8 + r] = psum[r];
  }
  __syncthreads();
  if (tid < 16) {
    float s = redsum[tid];
#pragma unroll
    for (int w2 = 1; w2 < 8; ++w2) s += redsum[w2 * 16 + tid];
    gsum[tid] = s;
  }
  __syncthreads();

  float* ATTb = ATT + ((size_t)b * LQ_ + q0) * LKV_;
#pragma unroll
  for (int r = 0; r < 8; ++r) {
    const int grow = hs * 8 + r;
    const float inv = 1.0f / gsum[grow];
#pragma unroll
    for (int t = 0; t < 8; ++t) {
      const int col = (w * 8 + t) * 16 + l15;
      const float p = acc[t][r] * inv;
      ATTb[(size_t)grow * LKV_ + col] = p;
      Ph[grow * LKV_ + col] = (_Float16)p;
    }
  }
  __syncthreads();

  v8f accO[4] = {};
  for (int kc = 0; kc < LKV_ / 32; ++kc) {
    for (int e = tid; e < 32 * D_; e += 256) {
      const int kv = e >> 9;
      const int d  = e & (D_ - 1);
      const float v = KVb[(size_t)(kc * 32 + kv) * D_ + d];
      const int qk = (kv >> 3) ^ ((d >> 2) & 3);
      KVT[d * 32 + qk * 8 + (kv & 7)] = (_Float16)v;
    }
    __syncthreads();

    V16U a;
    const _Float16* ap = Ph + l15 * LKV_ + kc * 32 + hs * 8;
    a.u[0] = *(const uint4*)(ap);
    a.u[1] = *(const uint4*)(ap + 16);
#pragma unroll
    for (int dt = 0; dt < 4; ++dt) {
      const int dcol = (w * 4 + dt) * 16 + l15;
      const int sw = (dcol >> 2) & 3;
      const uint4* rp = (const uint4*)(KVT + dcol * 32);
      V16U bm;
      bm.u[0] = rp[hs ^ sw];
      bm.u[1] = rp[(hs + 2) ^ sw];
      accO[dt] = WMMA_F16(a.h, bm.h, accO[dt]);
    }
    __syncthreads();
  }

  float* Ob = O + ((size_t)b * LQ_ + q0) * D_;
#pragma unroll
  for (int dt = 0; dt < 4; ++dt)
#pragma unroll
    for (int r = 0; r < 8; ++r)
      Ob[(size_t)(hs * 8 + r) * D_ + (w * 4 + dt) * 16 + l15] = accO[dt][r];
}

extern "C" void kernel_launch(void* const* d_in, const int* in_sizes, int n_in,
                              void* d_out, int out_size, void* d_ws, size_t ws_size,
                              hipStream_t stream) {
  const float* Q  = (const float*)d_in[0];
  const float* KV = (const float*)d_in[1];
  float* O   = (float*)d_out;
  float* ATT = O + (size_t)B_ * LQ_ * D_;   // tuple order: (output, attn_score)

  const size_t kvhBytes = (size_t)B_ * LKV_ * D_ * 2;  // f16 KV copy
  if (ws_size >= kvhBytes) {
    _Float16* KVh = (_Float16*)d_ws;
    const int nvec = (B_ * LKV_ * D_) / 8;             // 8 elems / thread
    cvt_f32_to_f16<<<nvec / 256, 256, 0, stream>>>(KV, KVh);
    attn_fused32<<<dim3(LQ_ / 32, B_), 256, 0, stream>>>(Q, KVh, O, ATT);
  } else {
    attn_fused<<<dim3(LQ_ / 16, B_), 256, 0, stream>>>(Q, KV, O, ATT);
  }
}
